// MultiHeadSelfAttention_78056735637855
// MI455X (gfx1250) — matrix operand order repaired and checked by reading, not compiled, not run on hardware
//
#include <hip/hip_runtime.h>
#include <hip/hip_bf16.h>
#include <stdint.h>

#define D_MODEL 1024
#define NHEADS  16
#define HEADDIM 64
#define SEQ     2048
#define BATCH   2
#define MROWS   (BATCH * SEQ)   // 4096

typedef __bf16 bf16_t;
typedef bf16_t v16bf __attribute__((ext_vector_type(16)));
typedef float  v8f   __attribute__((ext_vector_type(8)));

union Frag32  { uint4 u[2]; v16bf v; };
union Pack8   { uint4 u; bf16_t h[8]; };
union BfBits  { bf16_t b; unsigned short s; };
union V8fBits { v8f v; uint4 u[2]; };

__device__ __forceinline__ v16bf join16(uint4 lo, uint4 hi) {
  Frag32 f; f.u[0] = lo; f.u[1] = hi; return f.v;
}

__device__ __forceinline__ v8f wmma_bf16(v16bf a, v16bf b, v8f c) {
  // D(f32 16x16) = A(bf16 16x32) * B(bf16 32x16) + C
  return __builtin_amdgcn_wmma_f32_16x16x32_bf16(false, a, false, b, (short)0, c,
                                                 false, false);
}

// ---------------------------------------------------------------------------
// f32 -> bf16 conversion (row-major preserved), vectorized by 4
// ---------------------------------------------------------------------------
__global__ __launch_bounds__(256) void cvt_bf16_kernel(const float* __restrict__ src,
                                                       bf16_t* __restrict__ dst, int n) {
  int i = (blockIdx.x * 256 + threadIdx.x) * 4;
  if (i < n) {
    float4 f = *(const float4*)(src + i);
    BfBits b0, b1, b2, b3;
    b0.b = (bf16_t)f.x; b1.b = (bf16_t)f.y; b2.b = (bf16_t)f.z; b3.b = (bf16_t)f.w;
    ushort4 pk; pk.x = b0.s; pk.y = b1.s; pk.z = b2.s; pk.w = b3.s;
    *(ushort4*)(dst + i) = pk;
  }
}

// ---------------------------------------------------------------------------
// One-time tiled transpose + convert: W[K][N] f32 -> Wt[N][K] bf16.
// 64x64 tile through LDS; global reads float4-coalesced, writes uint4-coalesced.
// ---------------------------------------------------------------------------
#define TST 72
__global__ __launch_bounds__(256)
void cvt_transpose_kernel(const float* __restrict__ W, bf16_t* __restrict__ Wt,
                          int Kdim, int N) {
  __shared__ bf16_t tile[64 * TST];   // [n][k]
  const int t  = threadIdx.x;
  const int n0 = blockIdx.x * 64;
  const int k0 = blockIdx.y * 64;
#pragma unroll
  for (int j = 0; j < 4; ++j) {
    int e = t * 16 + j * 4;
    int kr = e >> 6, nc = e & 63;
    float4 f = *(const float4*)(W + (size_t)(k0 + kr) * N + n0 + nc);
    tile[(nc + 0) * TST + kr] = (bf16_t)f.x;
    tile[(nc + 1) * TST + kr] = (bf16_t)f.y;
    tile[(nc + 2) * TST + kr] = (bf16_t)f.z;
    tile[(nc + 3) * TST + kr] = (bf16_t)f.w;
  }
  __syncthreads();
#pragma unroll
  for (int j = 0; j < 2; ++j) {
    int e = t * 16 + j * 8;
    int nr = e >> 6, kc = e & 63;
    uint4 pk = *(const uint4*)(&tile[nr * TST + kc]);
    *(uint4*)(Wt + (size_t)(n0 + nr) * Kdim + k0 + kc) = pk;
  }
}

// ---------------------------------------------------------------------------
// Tiled bf16 GEMM: C[M,N] = A[M,K] * Bt[N,K]^T + bias[N]
//   A staged in LDS (128 rows x 128 K per stage, vector in/out).
//   B fragments loaded directly from global transposed weights (contiguous
//   16-lane / 16-bit B layout, L2-resident).
//   8 waves: wave w -> (wm = w>>1) 32-row band, (wn = w&1) 32-col half.
//   mode 0: scatter bf16 into Q/K [bh][t][d] and V transposed [bh][d][t]
//   mode 1: write f32 to outp [M,N]
// ---------------------------------------------------------------------------
#define AST 136   // 272B rows: 16B aligned, bank-spread
#define KSTEP 128

__global__ __launch_bounds__(256)
void gemm_bf16_kernel(const bf16_t* __restrict__ A, const bf16_t* __restrict__ Bt,
                      int Kdim, int N, const float* __restrict__ bias, int mode,
                      bf16_t* __restrict__ Qp, bf16_t* __restrict__ Kp,
                      bf16_t* __restrict__ Vp, float* __restrict__ outp) {
  __shared__ bf16_t As[128 * AST];

  const int t     = threadIdx.x;
  const int m0    = blockIdx.y * 128;
  const int n0    = blockIdx.x * 64;
  const int w     = t >> 5;
  const int lane  = t & 31;
  const int lrow  = lane & 15;
  const int lhalf = lane >> 4;
  const int wm    = w >> 1;   // 0..3
  const int wn    = w & 1;    // 0..1

  v8f acc[2][2];
  acc[0][0] = (v8f){}; acc[0][1] = (v8f){}; acc[1][0] = (v8f){}; acc[1][1] = (v8f){};

  for (int kk = 0; kk < Kdim; kk += KSTEP) {
    // ---- stage A tile: 128 rows x 128 K, all 16B vector traffic ----
#pragma unroll
    for (int j = 0; j < 8; ++j) {
      int e = t * 64 + j * 8;
      int r = e >> 7, c = e & 127;
      uint4 pk = *(const uint4*)(A + (size_t)(m0 + r) * Kdim + kk + c);
      *(uint4*)(&As[r * AST + c]) = pk;
    }
    __syncthreads();

#pragma unroll
    for (int kc = 0; kc < 4; ++kc) {
      v16bf afr[2], bfr[2];
#pragma unroll
      for (int mi = 0; mi < 2; ++mi) {
        int ar = wm * 32 + mi * 16 + lrow;
        uint4 lo = *(const uint4*)(&As[ar * AST + kc * 32 + lhalf * 16]);
        uint4 hi = *(const uint4*)(&As[ar * AST + kc * 32 + lhalf * 16 + 8]);
        afr[mi] = join16(lo, hi);
      }
#pragma unroll
      for (int ni = 0; ni < 2; ++ni) {
        int bn = n0 + wn * 32 + ni * 16 + lrow;      // lane column of B
        const bf16_t* bp = Bt + (size_t)bn * Kdim + kk + kc * 32 + lhalf * 16;
        uint4 lo = *(const uint4*)(bp);
        uint4 hi = *(const uint4*)(bp + 8);
        bfr[ni] = join16(lo, hi);
      }
#pragma unroll
      for (int mi = 0; mi < 2; ++mi)
#pragma unroll
        for (int ni = 0; ni < 2; ++ni)
          acc[mi][ni] = wmma_bf16(afr[mi], bfr[ni], acc[mi][ni]);
    }
    __syncthreads();
  }

  // ---- epilogue: C layout -> row = base + lhalf*8 + v, col = base + lrow ----
#pragma unroll
  for (int mi = 0; mi < 2; ++mi) {
#pragma unroll
    for (int ni = 0; ni < 2; ++ni) {
      int ncol = n0 + wn * 32 + ni * 16 + lrow;
      float bv = bias[ncol];
#pragma unroll
      for (int v = 0; v < 8; ++v) {
        int mrow = m0 + wm * 32 + mi * 16 + lhalf * 8 + v;
        float val = acc[mi][ni][v] + bv;
        if (mode == 0) {
          int sel = ncol >> 10, rem = ncol & 1023;
          int h = rem >> 6, d = rem & 63;
          int b = mrow >> 11, tq = mrow & 2047;
          size_t bh = (size_t)(b * NHEADS + h);
          if (sel == 0)      Qp[(bh * SEQ + tq) * HEADDIM + d] = (bf16_t)val;
          else if (sel == 1) Kp[(bh * SEQ + tq) * HEADDIM + d] = (bf16_t)val;
          else               Vp[bh * HEADDIM * SEQ + (size_t)d * SEQ + tq] = (bf16_t)val;
        } else {
          outp[(size_t)mrow * N + ncol] = val;
        }
      }
    }
  }
}

// ---------------------------------------------------------------------------
// Flash-style attention: block = 64 queries of one (b,h); 8 waves.
// Wave w: qt = w>>1 (16-query tile), half = w&1 (32-key / 32-dim half).
// K and V^T fragments stream straight from global (L2-resident); LDS only
// holds the transposed S spill, P, and the online-softmax stats.
// ---------------------------------------------------------------------------
#define SST 68   // S_t[key][query] stride (f32)
#define PST 72   // P[query][key] stride (bf16)

__global__ __launch_bounds__(256)
void attn_kernel(const bf16_t* __restrict__ Qg, const bf16_t* __restrict__ Kg,
                 const bf16_t* __restrict__ Vtg, bf16_t* __restrict__ ctx) {
  __shared__ float  S_s[64 * SST];     // transposed scores
  __shared__ bf16_t P_s[64 * PST];
  __shared__ float  m_s[64], l_s[64], a_s[64];

  const int t     = threadIdx.x;
  const int qb    = blockIdx.x;   // query block (64 queries)
  const int bh    = blockIdx.y;   // b*H + h
  const int w     = t >> 5;
  const int lane  = t & 31;
  const int lrow  = lane & 15;
  const int lhalf = lane >> 4;
  const int qt    = w >> 1;
  const int half  = w & 1;

  const bf16_t* Qp = Qg  + (size_t)bh * SEQ * HEADDIM;
  const bf16_t* Kp = Kg  + (size_t)bh * SEQ * HEADDIM;
  const bf16_t* Vt = Vtg + (size_t)bh * HEADDIM * SEQ;   // [d][t]

  if (t < 64) { m_s[t] = -1e30f; l_s[t] = 0.0f; }

  // Q fragments held in registers for the whole pass (A layout: 16x32 per kc)
  v16bf qf[2];
  {
    int qrow = qb * 64 + qt * 16 + lrow;
#pragma unroll
    for (int kc = 0; kc < 2; ++kc) {
      uint4 lo = *(const uint4*)(Qp + (size_t)qrow * HEADDIM + kc * 32 + lhalf * 16);
      uint4 hi = *(const uint4*)(Qp + (size_t)qrow * HEADDIM + kc * 32 + lhalf * 16 + 8);
      qf[kc] = join16(lo, hi);
    }
  }
  v8f o0 = (v8f){}, o1 = (v8f){};
  __syncthreads();

  for (int kb = 0; kb < SEQ; kb += 64) {
    // ---- S = Q K^T for this wave's two 16-key fragments ----
    v8f s0 = (v8f){}, s1 = (v8f){};
#pragma unroll
    for (int ni = 0; ni < 2; ++ni) {
      int key = kb + (half * 2 + ni) * 16 + lrow;   // B frag: lane col = key
#pragma unroll
      for (int kc = 0; kc < 2; ++kc) {
        uint4 lo = *(const uint4*)(Kp + (size_t)key * HEADDIM + kc * 32 + lhalf * 16);
        uint4 hi = *(const uint4*)(Kp + (size_t)key * HEADDIM + kc * 32 + lhalf * 16 + 8);
        v16bf kf = join16(lo, hi);
        if (ni == 0) s0 = wmma_bf16(qf[kc], kf, s0);
        else         s1 = wmma_bf16(qf[kc], kf, s1);
      }
    }
    // ---- spill S transposed: S_t[key][query]; 2x b128 per fragment ----
    {
      int rbase = qt * 16 + lhalf * 8;
      V8fBits sb0; sb0.v = s0;
      int c0 = (half * 2 + 0) * 16 + lrow;
      *(uint4*)(&S_s[c0 * SST + rbase])     = sb0.u[0];
      *(uint4*)(&S_s[c0 * SST + rbase + 4]) = sb0.u[1];
      V8fBits sb1; sb1.v = s1;
      int c1 = (half * 2 + 1) * 16 + lrow;
      *(uint4*)(&S_s[c1 * SST + rbase])     = sb1.u[0];
      *(uint4*)(&S_s[c1 * SST + rbase + 4]) = sb1.u[1];
    }
    __syncthreads();

    // ---- online softmax: one thread per query row (conflict-free reads) ----
    if (t < 64) {
      const float scale = 0.125f;   // 1/sqrt(64)
      float mold = m_s[t], lold = l_s[t];
      float mx = mold;
#pragma unroll 8
      for (int j = 0; j < 64; ++j) mx = fmaxf(mx, S_s[j * SST + t] * scale);
      float alpha = __expf(mold - mx);
      float lsum = lold * alpha;
#pragma unroll
      for (int j0 = 0; j0 < 64; j0 += 8) {
        Pack8 pk;
#pragma unroll
        for (int q = 0; q < 8; ++q) {
          float p = __expf(S_s[(j0 + q) * SST + t] * scale - mx);
          lsum += p;
          pk.h[q] = (bf16_t)p;
        }
        *(uint4*)(&P_s[t * PST + j0]) = pk.u;
      }
      m_s[t] = mx; l_s[t] = lsum; a_s[t] = alpha;
    }
    __syncthreads();

    // ---- rescale running O, then O += P * V (V^T streamed from global) ----
#pragma unroll
    for (int v = 0; v < 8; ++v) {
      float al = a_s[qt * 16 + lhalf * 8 + v];
      o0[v] *= al; o1[v] *= al;
    }
#pragma unroll
    for (int kc = 0; kc < 2; ++kc) {
      int pr = qt * 16 + lrow;
      uint4 lo = *(const uint4*)(&P_s[pr * PST + kc * 32 + lhalf * 16]);
      uint4 hi = *(const uint4*)(&P_s[pr * PST + kc * 32 + lhalf * 16 + 8]);
      v16bf pa = join16(lo, hi);
#pragma unroll
      for (int ni = 0; ni < 2; ++ni) {
        int d = (half * 2 + ni) * 16 + lrow;         // lane column of V^T
        const bf16_t* vp = Vt + (size_t)d * SEQ + kb + kc * 32 + lhalf * 16;
        uint4 blo = *(const uint4*)(vp);
        uint4 bhi = *(const uint4*)(vp + 8);
        v16bf vb = join16(blo, bhi);
        if (ni == 0) o0 = wmma_bf16(pa, vb, o0);
        else         o1 = wmma_bf16(pa, vb, o1);
      }
    }
    __syncthreads();
  }

  // ---- normalize and write context bf16 [B,T,D] (head-interleaved) ----
  int b = bh / NHEADS, h = bh % NHEADS;
#pragma unroll
  for (int v = 0; v < 8; ++v) {
    int r = qt * 16 + lhalf * 8 + v;
    float inv = 1.0f / l_s[r];
    int qidx = qb * 64 + r;
    size_t base = ((size_t)b * SEQ + qidx) * D_MODEL + h * HEADDIM;
    ctx[base + half * 32 + 0 * 16 + lrow] = (bf16_t)(o0[v] * inv);
    ctx[base + half * 32 + 1 * 16 + lrow] = (bf16_t)(o1[v] * inv);
  }
}

// ---------------------------------------------------------------------------
// Launch
// ---------------------------------------------------------------------------
extern "C" void kernel_launch(void* const* d_in, const int* in_sizes, int n_in,
                              void* d_out, int out_size, void* d_ws, size_t ws_size,
                              hipStream_t stream) {
  const float* x    = (const float*)d_in[0];
  const float* Wqkv = (const float*)d_in[1];
  const float* bqkv = (const float*)d_in[2];
  const float* Wout = (const float*)d_in[3];
  const float* bout = (const float*)d_in[4];
  float* out = (float*)d_out;

  char* ws = (char*)d_ws;
  size_t off = 0;
  auto alloc = [&](size_t bytes) -> void* {
    void* p = ws + off;
    off += (bytes + 255) & ~(size_t)255;
    return p;
  };
  const size_t nX    = (size_t)MROWS * D_MODEL;          // 4,194,304
  const size_t nWq   = (size_t)D_MODEL * 3 * D_MODEL;    // 3,145,728
  const size_t nWo   = (size_t)D_MODEL * D_MODEL;        // 1,048,576
  const size_t nHead = (size_t)BATCH * NHEADS * SEQ * HEADDIM;

  bf16_t* xb    = (bf16_t*)alloc(nX * 2);
  bf16_t* wqkvT = (bf16_t*)alloc(nWq * 2);   // [3072][1024]
  bf16_t* woutT = (bf16_t*)alloc(nWo * 2);   // [1024][1024]
  bf16_t* Qb    = (bf16_t*)alloc(nHead * 2); // [bh][t][d]
  bf16_t* Kb    = (bf16_t*)alloc(nHead * 2); // [bh][t][d]
  bf16_t* Vtb   = (bf16_t*)alloc(nHead * 2); // [bh][d][t]
  bf16_t* ctx   = (bf16_t*)alloc(nX * 2);

  // 1) convert x; transpose+convert weights (one-off, L2-resident afterwards)
  cvt_bf16_kernel<<<(int)(nX / 4 / 256), 256, 0, stream>>>(x, xb, (int)nX);
  cvt_transpose_kernel<<<dim3(3 * D_MODEL / 64, D_MODEL / 64), 256, 0, stream>>>(
      Wqkv, wqkvT, D_MODEL, 3 * D_MODEL);
  cvt_transpose_kernel<<<dim3(D_MODEL / 64, D_MODEL / 64), 256, 0, stream>>>(
      Wout, woutT, D_MODEL, D_MODEL);

  // 2) QKV projection: [4096,1024] @ [1024,3072] + bqkv -> Q/K/V^T bf16
  gemm_bf16_kernel<<<dim3(3 * D_MODEL / 64, MROWS / 128), 256, 0, stream>>>(
      xb, wqkvT, D_MODEL, 3 * D_MODEL, bqkv, 0, Qb, Kb, Vtb, nullptr);

  // 3) attention -> ctx bf16 [B,T,D]
  attn_kernel<<<dim3(SEQ / 64, BATCH * NHEADS), 256, 0, stream>>>(Qb, Kb, Vtb, ctx);

  // 4) output projection: ctx @ Wout + bout -> f32 out
  gemm_bf16_kernel<<<dim3(D_MODEL / 64, MROWS / 128), 256, 0, stream>>>(
      ctx, woutT, D_MODEL, D_MODEL, bout, 1, nullptr, nullptr, nullptr, out);
}
